// NoInputMassConserving_76355928588894
// MI455X (gfx1250) — compile-verified, matrix operand order checked by reading, not run on hardware
//
#include <hip/hip_runtime.h>
#include <hip/hip_bf16.h>

// Mass-conserving RNN scan for MI455X (gfx1250, wave32, WMMA).
// One workgroup (8 wave32 = 256 thr) owns 16 batch rows for all T steps.
// Wr staged once in LDS as f16 in WMMA-B fragment layout (K=0..95) plus an
// f32 tail (K=96..99); per-step GEMM = 3x v_wmma_f32_16x16x32_f16 +
// 1x v_wmma_f32_16x16x4_f32 per 16-wide N tile.

typedef __attribute__((ext_vector_type(16))) _Float16 v16h;
typedef __attribute__((ext_vector_type(8)))  _Float16 v8h;
typedef __attribute__((ext_vector_type(8)))  float    v8f;
typedef __attribute__((ext_vector_type(2)))  float    v2f;

#define THREADS 256
#define BT      16      // batch tile per workgroup
#define HS      32      // state size H
#define E1C     50
#define E2C     100
#define NCOL    1024    // H*H
#define NTILES  64      // NCOL/16

// LDS layout (bytes), total 325824 <= 327680 (320KB)
#define OFF_WR16   0         // [3][64][32][16] f16  B-frags K=0..95   196608
#define OFF_WRT    196608    // [64][32][2]     f32  B-frags K=96..99   16384
#define OFF_BR     212992    // [1024] f32                               4096
#define OFF_W1     217088    // [41*50] f32                              8200
#define OFF_B1     225288    // [50] f32                                  200
#define OFF_W2     225488    // [50*100] f32                            20000
#define OFF_B2     245488    // [100] f32                                 400
#define OFF_LOG    245888    // [16][1024] f32 logits (aliased h buf)   65536
#define OFF_CF32   311424    // [16][100] f32 conc                       6400
#define OFF_CF16   317824    // [16][104] f16 conc                       3328
#define OFF_CT0    321152    // [16][32] f32                             2048
#define OFF_CT1    323200    // [16][32] f32                             2048
#define OFF_XB     325248    // [16][9] f32                               576
#define SMEM_TOTAL 325824

__device__ __forceinline__ v8f wmma_f16(v16h a, v16h b, v8f c) {
  return __builtin_amdgcn_wmma_f32_16x16x32_f16(false, a, false, b,
                                                (short)0, c, false, false);
}
__device__ __forceinline__ v8f wmma_f32x4(v2f a, v2f b, v8f c) {
  return __builtin_amdgcn_wmma_f32_16x16x4_f32(false, a, false, b,
                                               (short)0, c, false, false);
}

// A-fragment (16x32 f16): lane<16 -> row=lane, K = {0..7,16..23}+kbase;
// lane>=16 -> row=lane-16, K = {8..15,24..31}+kbase.
__device__ __forceinline__ v16h load_a16(const _Float16* cf16, int lane, int ks) {
  const int m = lane & 15;
  const _Float16* p = cf16 + m * 104 + ks * 32 + ((lane & 16) ? 8 : 0);
  v8h lo = *(const v8h*)(p);
  v8h hi = *(const v8h*)(p + 16);
  v16h r;
#pragma unroll
  for (int h = 0; h < 8; ++h) { r[h] = lo[h]; r[8 + h] = hi[h]; }
  return r;
}

// B-fragment (32x16 f16) pre-swizzled: 16 contiguous halves per lane.
__device__ __forceinline__ v16h load_b16(const _Float16* wr16, int lane, int ks, int nt) {
  const _Float16* p = wr16 + (((ks * 64 + nt) * 32) + lane) * 16;
  v8h lo = *(const v8h*)(p);
  v8h hi = *(const v8h*)(p + 8);
  v16h r;
#pragma unroll
  for (int h = 0; h < 8; ++h) { r[h] = lo[h]; r[8 + h] = hi[h]; }
  return r;
}

__global__ __launch_bounds__(THREADS, 1)
void mass_rnn_scan(const float* __restrict__ init_state,
                   const float* __restrict__ xa,
                   const float* __restrict__ W1, const float* __restrict__ b1,
                   const float* __restrict__ W2, const float* __restrict__ b2,
                   const float* __restrict__ Wr, const float* __restrict__ br,
                   const int* __restrict__ nsteps,
                   float* __restrict__ out, int B, int Tfull) {
  extern __shared__ char smem[];
  _Float16* wr16 = (_Float16*)(smem + OFF_WR16);
  float*    wrt  = (float*)(smem + OFF_WRT);
  float*    brs  = (float*)(smem + OFF_BR);
  float*    w1s  = (float*)(smem + OFF_W1);
  float*    b1s  = (float*)(smem + OFF_B1);
  float*    w2s  = (float*)(smem + OFF_W2);
  float*    b2s  = (float*)(smem + OFF_B2);
  float*    logits = (float*)(smem + OFF_LOG);
  float*    hbuf   = (float*)(smem + OFF_LOG);   // alias: free during MLP1
  float*    cf32 = (float*)(smem + OFF_CF32);
  _Float16* cf16 = (_Float16*)(smem + OFF_CF16);
  float*    ct0  = (float*)(smem + OFF_CT0);
  float*    ct1  = (float*)(smem + OFF_CT1);
  float*    xb   = (float*)(smem + OFF_XB);

  const int tid  = threadIdx.x;
  const int lane = tid & 31;
  const int wave = tid >> 5;
  const int b0   = blockIdx.x * BT;
  const int nT   = nsteps[0];

  float* mo = out;                                 // [B, nT, 32]
  float* cg = out + (size_t)B * nT * HS;           // [B, nT+1, 32]

  // ---- one-time preload: Wr -> swizzled f16 B-fragments (K 0..95) ----
  for (int e = tid; e < 3 * 64 * 32; e += THREADS) {
    const int l  = e & 31;
    const int nt = (e >> 5) & 63;
    const int ks = e >> 11;
    const int n  = nt * 16 + (l & 15);
    const int kb = ks * 32 + ((l & 16) ? 8 : 0);
    _Float16* dst = wr16 + e * 16;
#pragma unroll
    for (int h = 0; h < 16; ++h) {
      const int k = kb + ((h < 8) ? h : (h + 8));
      dst[h] = (_Float16)Wr[(size_t)k * NCOL + n];
    }
  }
  // ---- f32 B-fragments for K tail 96..99 ----
  for (int e = tid; e < 64 * 32; e += THREADS) {
    const int l  = e & 31;
    const int nt = e >> 5;
    const int n  = nt * 16 + (l & 15);
    const int kb = 96 + ((l & 16) ? 2 : 0);
    wrt[e * 2 + 0] = Wr[(size_t)kb * NCOL + n];
    wrt[e * 2 + 1] = Wr[(size_t)(kb + 1) * NCOL + n];
  }
  for (int i = tid; i < NCOL; i += THREADS) brs[i] = br[i];
  for (int i = tid; i < 41 * E1C; i += THREADS) w1s[i] = W1[i];
  for (int i = tid; i < E1C; i += THREADS) b1s[i] = b1[i];
  for (int i = tid; i < E1C * E2C; i += THREADS) w2s[i] = W2[i];
  for (int i = tid; i < E2C; i += THREADS) b2s[i] = b2[i];
  for (int i = tid; i < BT * HS; i += THREADS) {
    const int m = i >> 5, hh = i & 31;
    const float v = init_state[(size_t)(b0 + m) * HS + hh];
    ct0[m * HS + hh] = v;
    cg[(size_t)(b0 + m) * (nT + 1) * HS + hh] = v;   // c[:,0,:]
  }
  __syncthreads();

  float* ct  = ct0;
  float* ctn = ct1;

  for (int t = 0; t < nT; ++t) {
    // ---- load x[t] for the 16 rows; prefetch x[t+1] ----
    for (int i = tid; i < BT * 9; i += THREADS) {
      const int m = i / 9, a = i % 9;
      const size_t base = ((size_t)(b0 + m) * Tfull + t) * 9 + a;
      xb[m * 9 + a] = xa[base];
      if (t + 1 < nT) __builtin_prefetch(xa + base + 9, 0, 1);
    }
    __syncthreads();

    // ---- MLP layer 1: h = relu([ct,x] @ W1 + b1) ----
    for (int i = tid; i < BT * E1C; i += THREADS) {
      const int m = i / E1C, e = i % E1C;
      float acc = b1s[e];
      const float* cm = ct + m * HS;
#pragma unroll
      for (int k = 0; k < HS; ++k) acc = fmaf(cm[k], w1s[k * E1C + e], acc);
      const float* xm = xb + m * 9;
#pragma unroll
      for (int a = 0; a < 9; ++a) acc = fmaf(xm[a], w1s[(HS + a) * E1C + e], acc);
      hbuf[m * 52 + e] = fmaxf(acc, 0.f);
    }
    __syncthreads();

    // ---- MLP layer 2: conc = relu(h @ W2 + b2) ----
    for (int i = tid; i < BT * E2C; i += THREADS) {
      const int m = i / E2C, e = i % E2C;
      float acc = b2s[e];
      const float* hm = hbuf + m * 52;
#pragma unroll
      for (int k = 0; k < E1C; ++k) acc = fmaf(hm[k], w2s[k * E2C + e], acc);
      const float r = fmaxf(acc, 0.f);
      cf32[m * E2C + e] = r;
      cf16[m * 104 + e] = (_Float16)r;
    }
    __syncthreads();

    // ---- big GEMM: logits[16,1024] = conc[16,100] @ Wr + br, via WMMA ----
    {
      const v16h A0 = load_a16(cf16, lane, 0);
      const v16h A1 = load_a16(cf16, lane, 1);
      const v16h A2 = load_a16(cf16, lane, 2);
      const float* pa = cf32 + (lane & 15) * E2C + 96 + ((lane & 16) ? 2 : 0);
      const v2f A3 = *(const v2f*)pa;
      const int nloc  = lane & 15;
      const int mbase = (lane & 16) ? 8 : 0;
#pragma unroll
      for (int jj = 0; jj < 8; ++jj) {
        const int nt = wave * 8 + jj;
        v8f c = {0.f, 0.f, 0.f, 0.f, 0.f, 0.f, 0.f, 0.f};
        c = wmma_f16(A0, load_b16(wr16, lane, 0, nt), c);
        c = wmma_f16(A1, load_b16(wr16, lane, 1, nt), c);
        c = wmma_f16(A2, load_b16(wr16, lane, 2, nt), c);
        const v2f Bt = *(const v2f*)(wrt + (nt * 32 + lane) * 2);
        c = wmma_f32x4(A3, Bt, c);
        const int n = nt * 16 + nloc;
        const float bias = brs[n];
#pragma unroll
        for (int v = 0; v < 8; ++v)
          logits[(mbase + v) * NCOL + n] = c[v] + bias;
      }
    }
    __syncthreads();

    // ---- row softmax over last axis (32-wide rows), one row per lane ----
    for (int r0 = tid; r0 < BT * HS; r0 += THREADS) {
      const int m = r0 >> 5, i = r0 & 31;
      float* row = logits + m * NCOL + i * HS;
      float mx = row[0];
#pragma unroll
      for (int j = 1; j < HS; ++j) mx = fmaxf(mx, row[j]);
      float s = 0.f;
#pragma unroll
      for (int j = 0; j < HS; ++j) { const float e = __expf(row[j] - mx); row[j] = e; s += e; }
      const float inv = 1.f / s;
#pragma unroll
      for (int j = 0; j < HS; ++j) row[j] *= inv;
    }
    __syncthreads();

    // ---- redistribution c_out[j] = sum_i ct[i]*r[i][j]; emit outputs ----
    for (int i0 = tid; i0 < BT * HS; i0 += THREADS) {
      const int m = i0 >> 5, j = i0 & 31;
      const float* cm = ct + m * HS;
      const float* rm = logits + m * NCOL + j;
      float acc = 0.f;
#pragma unroll
      for (int i = 0; i < HS; ++i) acc = fmaf(cm[i], rm[i * HS], acc);
      ctn[m * HS + j] = acc;
      const size_t gb = (size_t)(b0 + m);
      mo[(gb * nT + t) * HS + j] = acc;
      cg[(gb * (nT + 1) + t + 1) * HS + j] = 1.02f * acc - 0.01f;
    }
    __syncthreads();

    float* tmp = ct; ct = ctn; ctn = tmp;
  }
}

extern "C" void kernel_launch(void* const* d_in, const int* in_sizes, int n_in,
                              void* d_out, int out_size, void* d_ws, size_t ws_size,
                              hipStream_t stream) {
  const float* init_state = (const float*)d_in[0];
  const float* xa = (const float*)d_in[1];
  const float* W1 = (const float*)d_in[2];
  const float* b1 = (const float*)d_in[3];
  const float* W2 = (const float*)d_in[4];
  const float* b2 = (const float*)d_in[5];
  const float* Wr = (const float*)d_in[6];
  const float* br = (const float*)d_in[7];
  const int* nsteps = (const int*)d_in[8];

  const int B = in_sizes[0] / HS;                 // 4096
  const int Tfull = in_sizes[1] / (B * 9);        // 512 (xa time extent)

  dim3 grid(B / BT), block(THREADS);
  hipLaunchKernelGGL(mass_rnn_scan, grid, block, SMEM_TOTAL, stream,
                     init_state, xa, W1, b1, W2, b2, Wr, br, nsteps,
                     (float*)d_out, B, Tfull);
}
